// BahdanauAttention_35699768164501
// MI455X (gfx1250) — matrix operand order repaired and checked by reading, not compiled, not run on hardware
//
#include <hip/hip_runtime.h>
#include <hip/hip_bf16.h>
#include <stdint.h>

// ---------------------------------------------------------------------------
// Bahdanau attention, fused for MI455X (gfx1250, wave32, WMMA).
// Main GEMM (65536x512 @ 512x512) as bf16 WMMA / f32 accum:
//   - A/B fragments loaded as ds_load_b128 from LDS (contiguous k order: k = 16*hi + s)
//   - W1T slab staged with GLOBAL_LOAD_ASYNC_TO_LDS_B128, double-buffered,
//     synchronized with s_wait_asynccnt (ASYNCcnt) + one barrier per K-step.
// The 128 MiB score tensor is never materialized: tanh + V-dot fused in regs.
// ---------------------------------------------------------------------------

typedef __attribute__((ext_vector_type(16))) __bf16 v16bf;
typedef __attribute__((ext_vector_type(8)))  float  v8f;
typedef __attribute__((ext_vector_type(4)))  int    v4i;

#define UNITS 512
#define FEAT  512
#define DECD  512
#define BATCH 64
#define TSEQ  1024
#define ROWS  (BATCH * TSEQ)

#define BLK_M     64            // rows (b,t) per block
#define SA_STRIDE 520           // bf16 elems per sA row (512 + 8 pad)
#define SB_STRIDE 40            // bf16 elems per sB row (32 + 8 pad)
#define SA_DW     (SA_STRIDE / 2)
#define SB_DW     (SB_STRIDE / 2)
#define KSTEPS    (FEAT / 32)

union AFrag { uint4 q[2]; unsigned u[8]; v16bf v; };

__device__ __forceinline__ unsigned short bf16_rne(float x) {
  unsigned u = __float_as_uint(x);
  unsigned r = (u + 0x7fffu + ((u >> 16) & 1u)) >> 16;
  return (unsigned short)r;
}
__device__ __forceinline__ unsigned pack_bf16x2(float a, float b) {
  return (unsigned)bf16_rne(a) | ((unsigned)bf16_rne(b) << 16);
}
__device__ __forceinline__ float fast_tanh(float x) {
#if __has_builtin(__builtin_amdgcn_tanhf)
  return __builtin_amdgcn_tanhf(x);          // v_tanh_f32 on gfx1250
#elif __has_builtin(__builtin_amdgcn_tanh_f32)
  return __builtin_amdgcn_tanh_f32(x);
#else
  return tanhf(x);
#endif
}

// Async global->LDS 16B move (ASYNCcnt path); sync fallback if unavailable.
__device__ __forceinline__ void async_ld16(const unsigned* g, unsigned* l) {
#if __has_builtin(__builtin_amdgcn_global_load_async_to_lds_b128)
  __builtin_amdgcn_global_load_async_to_lds_b128(
      (__attribute__((address_space(1))) v4i*)g,
      (__attribute__((address_space(3))) v4i*)l, 0, 0);
#else
  *(uint4*)l = *(const uint4*)g;
#endif
}
__device__ __forceinline__ void wait_async0() {
#if __has_builtin(__builtin_amdgcn_s_wait_asynccnt)
  __builtin_amdgcn_s_wait_asynccnt(0);
#elif defined(__gfx1250__)
  asm volatile("s_wait_asynccnt 0x0" ::: "memory");
#endif
}

// --- Kernel 1: W1 (fp32 [F][U]) -> W1T (bf16 [U][F]) -------------------------
__global__ __launch_bounds__(256) void bah_w1t_kernel(
    const float* __restrict__ W1, unsigned short* __restrict__ w1t) {
  __shared__ float tile[32][33];
  const int tx = threadIdx.x & 31, ty = threadIdx.x >> 5;
  const int f0 = blockIdx.x * 32, u0 = blockIdx.y * 32;
#pragma unroll
  for (int r = ty; r < 32; r += 8)
    tile[r][tx] = W1[(size_t)(f0 + r) * UNITS + u0 + tx];
  __syncthreads();
#pragma unroll
  for (int r = ty; r < 32; r += 8)
    w1t[(size_t)(u0 + r) * FEAT + f0 + tx] = bf16_rne(tile[tx][r]);
}

// --- Kernel 2: pd[b][u] = dec[b] @ W2 + b2 ----------------------------------
__global__ __launch_bounds__(256) void bah_projdec_kernel(
    const float* __restrict__ dec, const float* __restrict__ W2,
    const float* __restrict__ b2, float* __restrict__ pd) {
  __shared__ float sdec[DECD];
  const int b = blockIdx.x, tid = threadIdx.x;
  for (int i = tid; i < DECD; i += 256) sdec[i] = dec[b * DECD + i];
  __syncthreads();
  float a0 = 0.f, a1 = 0.f;
  for (int d = 0; d < DECD; ++d) {
    float dv = sdec[d];
    a0 += dv * W2[(size_t)d * UNITS + tid];
    a1 += dv * W2[(size_t)d * UNITS + tid + 256];
  }
  pd[b * UNITS + tid]       = a0 + b2[tid];
  pd[b * UNITS + tid + 256] = a1 + b2[tid + 256];
}

// --- Kernel 3: fused  logits = tanh(enc@W1 + b1 + pd) @ V + bV --------------
// Block: 256 thr (8 waves), 64 rows x 512 units. Wave w owns n in [64w,64w+64).
__global__ __launch_bounds__(256) void bah_score_kernel(
    const float* __restrict__ enc, const unsigned* __restrict__ w1t,
    const float* __restrict__ pd, const float* __restrict__ b1,
    const float* __restrict__ Vv, const float* __restrict__ bV,
    float* __restrict__ logits) {
  extern __shared__ unsigned char smem[];
  unsigned* sA  = (unsigned*)smem;                  // BLK_M x SA_DW dwords
  unsigned* sB0 = sA + BLK_M * SA_DW;               // UNITS x SB_DW dwords
  unsigned* sB1 = sB0 + UNITS * SB_DW;              // UNITS x SB_DW dwords
  float* sLog   = (float*)(sB1 + UNITS * SB_DW);    // BLK_M floats

  const int tid  = threadIdx.x;
  const int wave = tid >> 5;
  const int lane = tid & 31;
  const int lo   = lane & 15;
  const int hi   = lane >> 4;
  const int row0 = blockIdx.x * BLK_M;              // global (b,t) row
  const int b    = row0 >> 10;

  // Issue async staging of W1T slab ks into buf: rows n=0..511, 16 dwords each.
  auto issue_slab = [&](int ks, unsigned* buf) {
#pragma unroll
    for (int r = 0; r < 8; ++r) {
      int i = tid + r * 256;                 // 0..2047 uint4 units
      int n = i >> 2;
      int q = (i & 3) * 4;                   // dword offset within row
      async_ld16(w1t + (size_t)n * (FEAT / 2) + ks * 16 + q,
                 buf + n * SB_DW + q);
    }
  };

  issue_slab(0, sB0);                        // prologue: slab 0 in flight

  // Stage A panel: enc[row0 : row0+64][0:512] fp32 -> bf16 (RNE) in LDS.
  {
    const float2* encp = (const float2*)enc + (size_t)row0 * (FEAT / 2);
    for (int i = tid; i < BLK_M * (FEAT / 2); i += 256) {
      int r = i >> 8;                        // 256 float2 per row
      int c = i & 255;
      float2 f = encp[(size_t)r * (FEAT / 2) + c];
      sA[r * SA_DW + c] = pack_bf16x2(f.x, f.y);
    }
  }
  if (tid < BLK_M) sLog[tid] = 0.f;

  v8f acc[4][4];
  const v8f vzero = {0.f, 0.f, 0.f, 0.f, 0.f, 0.f, 0.f, 0.f};
#pragma unroll
  for (int mt = 0; mt < 4; ++mt)
#pragma unroll
    for (int nt = 0; nt < 4; ++nt) acc[mt][nt] = vzero;

  const int n0base = wave * 64;

  for (int ks = 0; ks < KSTEPS; ++ks) {
    unsigned* sBc = (ks & 1) ? sB1 : sB0;
    unsigned* sBn = (ks & 1) ? sB0 : sB1;
    wait_async0();          // my slab-ks async loads landed
    __syncthreads();        // everyone's slab-ks visible; buffer sBn released
    if (ks + 1 < KSTEPS) issue_slab(ks + 1, sBn);

    // B fragments: k = 2v+pos+16*hi -> 8 contiguous dwords at +8*hi (2x b128).
    AFrag bf[4];
#pragma unroll
    for (int nt = 0; nt < 4; ++nt) {
      const uint4* p =
          (const uint4*)(sBc + (n0base + nt * 16 + lo) * SB_DW + 8 * hi);
      bf[nt].q[0] = p[0];
      bf[nt].q[1] = p[1];
    }
#pragma unroll
    for (int mt = 0; mt < 4; ++mt) {
      // A fragment: k = 2v+pos+16*hi -> 8 contiguous dwords at +8*hi (2x b128)
      AFrag af;
      const unsigned* pa = sA + (mt * 16 + lo) * SA_DW + ks * 16 + 8 * hi;
      af.q[0] = *(const uint4*)pa;
      af.q[1] = *(const uint4*)(pa + 4);
#pragma unroll
      for (int nt = 0; nt < 4; ++nt)
        acc[mt][nt] = __builtin_amdgcn_wmma_f32_16x16x32_bf16(
            false, af.v, false, bf[nt].v, (short)0, acc[mt][nt], false, false);
    }
  }

  // Epilogue: tanh(acc + b1 + pd) * V, reduce over this wave's 64 columns.
  float addn[4], vn[4];
#pragma unroll
  for (int nt = 0; nt < 4; ++nt) {
    int n = n0base + nt * 16 + lo;
    addn[nt] = b1[n] + pd[b * UNITS + n];
    vn[nt]   = Vv[n];
  }
#pragma unroll
  for (int mt = 0; mt < 4; ++mt) {
#pragma unroll
    for (int r = 0; r < 8; ++r) {      // C layout: m = 16*mt + r + 8*hi, n = lo
      float s = 0.f;
#pragma unroll
      for (int nt = 0; nt < 4; ++nt)
        s += fast_tanh(acc[mt][nt][r] + addn[nt]) * vn[nt];
      s += __shfl_xor(s, 1, 16);
      s += __shfl_xor(s, 2, 16);
      s += __shfl_xor(s, 4, 16);
      s += __shfl_xor(s, 8, 16);
      if (lo == 0) atomicAdd(&sLog[mt * 16 + 8 * hi + r], s);
    }
  }
  __syncthreads();
  if (tid < BLK_M) logits[row0 + tid] = sLog[tid] + bV[0];
}

// --- Kernel 4: softmax over T per batch -------------------------------------
__global__ __launch_bounds__(256) void bah_softmax_kernel(
    const float* __restrict__ logits, float* __restrict__ wout) {
  __shared__ float redm[8], reds[8];
  const int b = blockIdx.x, tid = threadIdx.x;
  const int wave = tid >> 5;
  const float* lb = logits + b * TSEQ;
  float l[4], m = -INFINITY;
#pragma unroll
  for (int j = 0; j < 4; ++j) { l[j] = lb[tid + j * 256]; m = fmaxf(m, l[j]); }
#pragma unroll
  for (int off = 16; off >= 1; off >>= 1) m = fmaxf(m, __shfl_xor(m, off, 32));
  if ((tid & 31) == 0) redm[wave] = m;
  __syncthreads();
  float M = redm[0];
#pragma unroll
  for (int w = 1; w < 8; ++w) M = fmaxf(M, redm[w]);
  float e[4], s = 0.f;
#pragma unroll
  for (int j = 0; j < 4; ++j) { e[j] = __expf(l[j] - M); s += e[j]; }
#pragma unroll
  for (int off = 16; off >= 1; off >>= 1) s += __shfl_xor(s, off, 32);
  if ((tid & 31) == 0) reds[wave] = s;
  __syncthreads();
  float S = 0.f;
#pragma unroll
  for (int w = 0; w < 8; ++w) S += reds[w];
  float inv = 1.f / S;
#pragma unroll
  for (int j = 0; j < 4; ++j) wout[b * TSEQ + tid + j * 256] = e[j] * inv;
}

// --- Kernel 5: context[b][f] = sum_t w[b][t] * enc[b][t][f] ------------------
__global__ __launch_bounds__(256) void bah_context_kernel(
    const float* __restrict__ enc, const float* __restrict__ w,
    float* __restrict__ ctx) {
  __shared__ float sw[TSEQ];
  const int b = blockIdx.y, chunk = blockIdx.x, tid = threadIdx.x;
  const float* wb = w + b * TSEQ;
  for (int i = tid; i < TSEQ; i += 256) sw[i] = wb[i];
  __syncthreads();
  const int f = chunk * 256 + tid;
  const float* eb = enc + (size_t)b * TSEQ * FEAT + f;
  float a0 = 0.f, a1 = 0.f, a2 = 0.f, a3 = 0.f;
  for (int t = 0; t < TSEQ; t += 4) {
    if (t + 16 < TSEQ)
      __builtin_prefetch(eb + (size_t)(t + 16) * FEAT, 0, 1);
    a0 += sw[t + 0] * eb[(size_t)(t + 0) * FEAT];
    a1 += sw[t + 1] * eb[(size_t)(t + 1) * FEAT];
    a2 += sw[t + 2] * eb[(size_t)(t + 2) * FEAT];
    a3 += sw[t + 3] * eb[(size_t)(t + 3) * FEAT];
  }
  ctx[b * FEAT + f] = (a0 + a1) + (a2 + a3);
}

// ---------------------------------------------------------------------------
extern "C" void kernel_launch(void* const* d_in, const int* in_sizes, int n_in,
                              void* d_out, int out_size, void* d_ws,
                              size_t ws_size, hipStream_t stream) {
  (void)in_sizes; (void)n_in; (void)out_size; (void)ws_size;
  const float* enc = (const float*)d_in[0];
  const float* dec = (const float*)d_in[1];
  const float* W1  = (const float*)d_in[2];
  const float* b1  = (const float*)d_in[3];
  const float* W2  = (const float*)d_in[4];
  const float* b2  = (const float*)d_in[5];
  const float* Vv  = (const float*)d_in[6];
  const float* bV  = (const float*)d_in[7];

  char* ws = (char*)d_ws;
  unsigned short* w1t = (unsigned short*)ws;                    // 512 KB
  float* pd     = (float*)(ws + 512 * 1024);                    // 128 KB
  float* logits = (float*)(ws + 512 * 1024 + 128 * 1024);       // 256 KB

  float* ctx  = (float*)d_out;                                  // [64][512]
  float* attw = (float*)d_out + BATCH * FEAT;                   // [64][1024]

  bah_w1t_kernel<<<dim3(16, 16), 256, 0, stream>>>(W1, w1t);
  bah_projdec_kernel<<<BATCH, 256, 0, stream>>>(dec, W2, b2, pd);

  size_t smain =
      (size_t)(BLK_M * SA_DW + 2 * UNITS * SB_DW) * 4 + BLK_M * 4;
  bah_score_kernel<<<ROWS / BLK_M, 256, smain, stream>>>(
      enc, (const unsigned*)w1t, pd, b1, Vv, bV, logits);

  bah_softmax_kernel<<<BATCH, 256, 0, stream>>>(logits, attw);
  bah_context_kernel<<<dim3(2, BATCH), 256, 0, stream>>>(enc, attw, ctx);
}
